// ARTTransformerBlock_27058293965551
// MI455X (gfx1250) — matrix operand order repaired and checked by reading, not compiled, not run on hardware
//
#include <hip/hip_runtime.h>
#include <hip/hip_bf16.h>

// ---------------------------------------------------------------------------
// ARTTransformerBlock on MI455X (gfx1250): bf16 WMMA GEMMs + fused epilogues,
// async global->LDS staging (ASYNCcnt) for the hot tiles.
// B=8, H=W=128, C=180, NH=6, hd=30, G=8, N=64, T=131072 tokens.
// ---------------------------------------------------------------------------

typedef unsigned short u16;
typedef __attribute__((ext_vector_type(16))) __bf16 v16bf;
typedef __attribute__((ext_vector_type(8)))  float  v8f;

#define TOKENS   131072          // B*L
#define C_DIM    180
#define QKV_DIM  540
#define FF_DIM   720
#define NWIN     2048            // B * (128/8) * (128/8)

union BFV { v16bf v; u16 u[16]; };

__device__ __forceinline__ u16 f32_to_bf16(float f) {
    union { float f; unsigned u; } x; x.f = f;
    unsigned r = x.u + 0x7fffu + ((x.u >> 16) & 1u);   // round-to-nearest-even
    return (u16)(r >> 16);
}

__device__ __forceinline__ v8f wmma_bf16(v16bf a, v16bf b, v8f c) {
    return __builtin_amdgcn_wmma_f32_16x16x32_bf16(
        /*neg_a=*/false, a, /*neg_b=*/false, b,
        /*c_mod=*/(short)0, c, /*reuse_a=*/false, /*reuse_b=*/false);
}

// LDS byte offset of a __shared__ object (flat LDS addr: low 32 bits = offset)
__device__ __forceinline__ unsigned lds_off(const void* p) {
    return (unsigned)(size_t)p;
}

// Async global->LDS: GVS mode (saddr base + 32-bit unsigned vgpr byte offset)
__device__ __forceinline__ void async_ld_b32(unsigned lds, unsigned goff,
                                             const void* saddr) {
    asm volatile("global_load_async_to_lds_b32 %0, %1, %2"
                 :: "v"(lds), "v"(goff), "s"(saddr) : "memory");
}
__device__ __forceinline__ void async_ld_b64(unsigned lds, unsigned goff,
                                             const void* saddr) {
    asm volatile("global_load_async_to_lds_b64 %0, %1, %2"
                 :: "v"(lds), "v"(goff), "s"(saddr) : "memory");
}
__device__ __forceinline__ void wait_async0() {
    asm volatile("s_wait_asynccnt 0x0" ::: "memory");
}

// A fragment (16x32 bf16): lane holds row m = mrow + lane%16; element i holds
// k = 16*(lane/16)+i. Tile is [M][ld] in LDS.
__device__ __forceinline__ v16bf lds_load_a(const u16* base, int mrow, int ld) {
    int lane = threadIdx.x & 31;
    const u16* p = base + (mrow + (lane & 15)) * ld;
    int kh = (lane >> 4) << 4;
    BFV f;
#pragma unroll
    for (int i = 0; i < 16; ++i) f.u[i] = p[kh + i];
    return f.v;
}

// B fragment (32x16 bf16) from a TRANSPOSED tile [N][ld]: lane holds column
// n = ncol + lane%16; elements i hold k = 16*(lane/16)+i (contiguous 32B read).
__device__ __forceinline__ v16bf lds_load_b(const u16* base, int ncol, int ld) {
    int lane = threadIdx.x & 31;
    const u16* p = base + (ncol + (lane & 15)) * ld + ((lane >> 4) << 4);
    BFV f;
#pragma unroll
    for (int i = 0; i < 16; ++i) f.u[i] = p[i];
    return f.v;
}

__device__ __forceinline__ float gelu_exact(float x) {
    return 0.5f * x * (1.0f + erff(x * 0.70710678118654752f));
}

// ---------------------------------------------------------------------------
// 1) dynamic relative-position-bias MLP: 225 offset rows x 11 features
// ---------------------------------------------------------------------------
__device__ __forceinline__ void ln_relu11(const float* t, float* u,
                                          const float* g, const float* b) {
    float m = 0.f;
#pragma unroll
    for (int i = 0; i < 11; ++i) m += t[i];
    m *= (1.f / 11.f);
    float v = 0.f;
#pragma unroll
    for (int i = 0; i < 11; ++i) { float d = t[i] - m; v += d * d; }
    v *= (1.f / 11.f);
    float rs = rsqrtf(v + 1e-5f);
#pragma unroll
    for (int i = 0; i < 11; ++i) {
        float y = (t[i] - m) * rs * g[i] + b[i];
        u[i] = y > 0.f ? y : 0.f;
    }
}

__device__ __forceinline__ void matvec11(const float* u, const float* w,
                                         const float* b, float* t) {
#pragma unroll
    for (int j = 0; j < 11; ++j) {
        float s = b[j];
#pragma unroll
        for (int i = 0; i < 11; ++i) s += u[i] * w[i * 11 + j];
        t[j] = s;
    }
}

__global__ void pos_mlp_kernel(
    const float* __restrict__ ppw, const float* __restrict__ ppb,
    const float* __restrict__ g1, const float* __restrict__ b1,
    const float* __restrict__ w1, const float* __restrict__ bb1,
    const float* __restrict__ g2, const float* __restrict__ b2,
    const float* __restrict__ w2, const float* __restrict__ bb2,
    const float* __restrict__ g3, const float* __restrict__ b3,
    const float* __restrict__ w3, const float* __restrict__ bb3,
    float* __restrict__ posT) {
    int r = blockIdx.x * blockDim.x + threadIdx.x;
    if (r >= 225) return;
    float bh = (float)(r / 15) - 7.f;
    float bw = (float)(r % 15) - 7.f;
    float t[11], u[11];
#pragma unroll
    for (int j = 0; j < 11; ++j) t[j] = bh * ppw[j] + bw * ppw[11 + j] + ppb[j];
    ln_relu11(t, u, g1, b1);
    matvec11(u, w1, bb1, t);
    ln_relu11(t, u, g2, b2);
    matvec11(u, w2, bb2, t);
    ln_relu11(t, u, g3, b3);
#pragma unroll
    for (int h = 0; h < 6; ++h) {
        float s = bb3[h];
#pragma unroll
        for (int i = 0; i < 11; ++i) s += u[i] * w3[i * 6 + h];
        posT[r * 6 + h] = s;
    }
}

// rpb[h][n][m]: n=query, m=key within an 8x8 window
__global__ void rpb_scatter_kernel(const float* __restrict__ posT,
                                   float* __restrict__ rpb) {
    int i = blockIdx.x * 256 + threadIdx.x;
    if (i >= 6 * 4096) return;
    int h = i >> 12, nm = i & 4095, n = nm >> 6, m = nm & 63;
    int dh = (n >> 3) - (m >> 3) + 7;
    int dw = (n & 7) - (m & 7) + 7;
    rpb[i] = posT[(dh * 15 + dw) * 6 + h];
}

// ---------------------------------------------------------------------------
// 2) LayerNorm (one wave32 per token, C=180) -> bf16; optional window partition
// ---------------------------------------------------------------------------
__global__ __launch_bounds__(256) void ln_kernel(
    const float* __restrict__ x, const float* __restrict__ g,
    const float* __restrict__ b, u16* __restrict__ out, int windowed) {
    int wave = threadIdx.x >> 5, lane = threadIdx.x & 31;
    int tok = blockIdx.x * 8 + wave;           // destination index
    int src;
    if (windowed) {
        int w = tok >> 6, n = tok & 63;
        int bi = w >> 8, wh = (w >> 4) & 15, ww = w & 15;
        int row = wh * 8 + (n >> 3), col = ww * 8 + (n & 7);
        src = bi * 16384 + row * 128 + col;
    } else {
        src = tok;
    }
    const float* xp = x + (size_t)src * C_DIM;
    float vals[6], s = 0.f, s2 = 0.f;
#pragma unroll
    for (int i = 0; i < 6; ++i) {
        int c = lane + i * 32;
        float v = (c < C_DIM) ? xp[c] : 0.f;
        vals[i] = v; s += v; s2 += v * v;
    }
#pragma unroll
    for (int off = 16; off; off >>= 1) {
        s  += __shfl_xor(s,  off, 32);
        s2 += __shfl_xor(s2, off, 32);
    }
    float mean = s * (1.f / C_DIM);
    float var  = s2 * (1.f / C_DIM) - mean * mean;
    float rstd = rsqrtf(var + 1e-5f);
    u16* op = out + (size_t)tok * C_DIM;
#pragma unroll
    for (int i = 0; i < 6; ++i) {
        int c = lane + i * 32;
        if (c < C_DIM) op[c] = f32_to_bf16((vals[i] - mean) * rstd * g[c] + b[c]);
    }
}

// ---------------------------------------------------------------------------
// 3) generic bf16 WMMA GEMM, 64x64 block tile, K staged in 32-chunks in LDS.
//    A tile staged with GLOBAL_LOAD_ASYNC_TO_LDS_B64 (ASYNCcnt), B tile
//    converted f32->bf16 synchronously (transposed [n][k] layout).
//    8 waves arranged 2(M) x 4(N); each wave owns a 32x16 output tile.
//    EPI: 0 = bias -> bf16, 1 = bias+GELU -> bf16, 2 = bias+residual -> f32.
//    Requires K % 4 == 0 and M % 64 == 0 (true for all call sites).
// ---------------------------------------------------------------------------
template <int EPI>
__global__ __launch_bounds__(256) void gemm_bf16(
    const u16* __restrict__ A,       // [M][K] bf16
    const float* __restrict__ Bw,    // [K][N] f32 weights
    const float* __restrict__ bias,  // [N]
    const float* __restrict__ resid, // [M][N] f32 (EPI==2)
    u16* __restrict__ out_bf, float* __restrict__ out_f,
    int M, int N, int K) {
    const int tilesN = (N + 63) >> 6;
    const int m0 = (blockIdx.x / tilesN) << 6;
    const int n0 = (blockIdx.x % tilesN) << 6;

    __shared__ u16 As[64 * 36];      // [m][k], stride 36 (72B rows, 8B aligned)
    __shared__ u16 Bs[64 * 36];      // [n][k] transposed

    const int tid = threadIdx.x;
    const int wave = tid >> 5, lane = tid & 31;
    const int wm = wave >> 2, wn = wave & 3;

    v8f acc0{}, acc1{};

    for (int kt = 0; kt < K; kt += 32) {
        // --- A tile: async global->LDS, 8 B64 units per row (4 bf16 each) ---
        for (int i = tid; i < 64 * 8; i += 256) {       // exactly 2 iterations
            int r = i >> 3, u = (i & 7) << 2;           // u16 column 0,4,...,28
            int gk = kt + u;
            unsigned lds = lds_off(&As[r * 36 + u]);
            if (gk + 3 < K || gk < K) {                 // K%4==0 -> gk<K suffices
                unsigned goff = (unsigned)(((size_t)(m0 + r) * K + gk) * 2u);
                async_ld_b64(lds, goff, A);
            } else {
                *(unsigned*)&As[r * 36 + u]       = 0u; // zero-pad OOB K
                *(unsigned*)(&As[r * 36 + u] + 2) = 0u;
            }
        }
        // prefetch next weight K-chunk while converting this one
        if (kt + 32 < K)
            __builtin_prefetch(&Bw[(size_t)(kt + 32) * N + n0], 0, 1);
        // --- B tile: f32 -> bf16, transposed [n][k] ---
        for (int i = tid; i < 64 * 32; i += 256) {
            int n = i >> 5, c = i & 31;
            int gn = n0 + n, gk = kt + c;
            u16 v = 0;
            if (gn < N && gk < K) v = f32_to_bf16(Bw[(size_t)gk * N + gn]);
            Bs[n * 36 + c] = v;
        }
        wait_async0();
        __syncthreads();
        v16bf bf = lds_load_b(Bs, wn * 16, 36);
        v16bf a0 = lds_load_a(As, wm * 32, 36);
        v16bf a1 = lds_load_a(As, wm * 32 + 16, 36);
        acc0 = wmma_bf16(a0, bf, acc0);
        acc1 = wmma_bf16(a1, bf, acc1);
        __syncthreads();
    }

    int gn = n0 + wn * 16 + (lane & 15);
    int rbase = m0 + wm * 32 + ((lane >> 4) << 3);
    if (gn < N) {
        float bv = bias[gn];
#pragma unroll
        for (int r = 0; r < 8; ++r) {
            int gm0 = rbase + r, gm1 = gm0 + 16;
            float v0 = acc0[r] + bv, v1 = acc1[r] + bv;
            if constexpr (EPI == 0) {
                if (gm0 < M) out_bf[(size_t)gm0 * N + gn] = f32_to_bf16(v0);
                if (gm1 < M) out_bf[(size_t)gm1 * N + gn] = f32_to_bf16(v1);
            } else if constexpr (EPI == 1) {
                if (gm0 < M) out_bf[(size_t)gm0 * N + gn] = f32_to_bf16(gelu_exact(v0));
                if (gm1 < M) out_bf[(size_t)gm1 * N + gn] = f32_to_bf16(gelu_exact(v1));
            } else {
                if (gm0 < M) out_f[(size_t)gm0 * N + gn] = v0 + resid[(size_t)gm0 * N + gn];
                if (gm1 < M) out_f[(size_t)gm1 * N + gn] = v1 + resid[(size_t)gm1 * N + gn];
            }
        }
    }
}

// ---------------------------------------------------------------------------
// 4) window attention: one block per window (64 tokens), loop over 6 heads.
//    q/k staged via GLOBAL_LOAD_ASYNC_TO_LDS_B32 (15 dwords per row = hd 30),
//    v staged synchronously (needs transpose). Pads pre-zeroed once.
//    S = (Q K^T)*scale + rpb ; softmax ; O = P V ; window-reverse on store.
// ---------------------------------------------------------------------------
__global__ __launch_bounds__(256) void attn_kernel(
    const u16* __restrict__ qkv,   // [Tw][540] bf16, window order
    const float* __restrict__ rpb, // [6][64][64]
    u16* __restrict__ out) {       // [T][180] bf16, token order
    const int w = blockIdx.x;
    const int tid = threadIdx.x, lane = tid & 31, wave = tid >> 5;
    const float scale = 0.18257418583505536f;   // 30^-0.5

    __shared__ u16 qs[64 * 36];    // [query][d]
    __shared__ u16 ks[64 * 36];    // [key][d]  (== B^T for QK^T)
    __shared__ u16 vt[32 * 72];    // [d][key]  (== B^T for PV)
    __shared__ float S[64 * 64];
    __shared__ u16 P[64 * 72];     // softmax probs, bf16 A-matrix

    // pre-zero the d=30,31 pad columns (asyncs never write them)
    if (tid < 64) {
        qs[tid * 36 + 30] = 0; qs[tid * 36 + 31] = 0;
        ks[tid * 36 + 30] = 0; ks[tid * 36 + 31] = 0;
    }

    for (int h = 0; h < 6; ++h) {
        // --- async q/k: 64 rows x 15 dwords (d = 0..29, dword-paired) ---
        for (int i = tid; i < 64 * 15; i += 256) {
            int n = i / 15, dd = (i % 15) * 2;          // d = dd, dd+1
            unsigned goff =
                (unsigned)((((size_t)(w * 64 + n)) * QKV_DIM + h * 30 + dd) * 2u);
            async_ld_b32(lds_off(&qs[n * 36 + dd]), goff, qkv);
            async_ld_b32(lds_off(&ks[n * 36 + dd]), goff + 360u, qkv); // +180 cols
        }
        // --- v: synchronous transpose into vt[d][key], zero pads d=30,31 ---
        for (int i = tid; i < 64 * 32; i += 256) {
            int n = i >> 5, d = i & 31;
            u16 vv = 0;
            if (d < 30)
                vv = qkv[(size_t)(w * 64 + n) * QKV_DIM + h * 30 + d + 360];
            vt[d * 72 + n] = vv;
        }
        wait_async0();
        __syncthreads();

        // S tiles: 4x4 of 16x16, K=32 (hd padded); each wave does 2 tiles.
        for (int t = wave; t < 16; t += 8) {
            int ti = t >> 2, tj = t & 3;
            v16bf a = lds_load_a(qs, ti * 16, 36);
            v16bf bf = lds_load_b(ks, tj * 16, 36);
            v8f acc{};
            acc = wmma_bf16(a, bf, acc);
            int n_ = tj * 16 + (lane & 15);
            const float* rp = rpb + ((size_t)h << 12);
#pragma unroll
            for (int r = 0; r < 8; ++r) {
                int m_ = ti * 16 + r + ((lane >> 4) << 3);
                S[m_ * 64 + n_] = acc[r] * scale + rp[m_ * 64 + n_];
            }
        }
        __syncthreads();

        // softmax: one thread per query row
        if (tid < 64) {
            float mx = -1e30f;
            for (int c = 0; c < 64; ++c) mx = fmaxf(mx, S[tid * 64 + c]);
            float sum = 0.f;
            for (int c = 0; c < 64; ++c) {
                float e = expf(S[tid * 64 + c] - mx);
                S[tid * 64 + c] = e;
                sum += e;
            }
            float inv = 1.f / sum;
            for (int c = 0; c < 64; ++c)
                P[tid * 72 + c] = f32_to_bf16(S[tid * 64 + c] * inv);
        }
        __syncthreads();

        // O = P @ V : tiles 4(M) x 2(Npad32); one tile per wave, K=64 in 2 steps
        {
            int ti = wave >> 1, tj = wave & 1;
            v8f acc{};
#pragma unroll
            for (int k0 = 0; k0 < 64; k0 += 32) {
                v16bf a = lds_load_a(P + k0, ti * 16, 72);
                v16bf bf = lds_load_b(vt + k0, tj * 16, 72);
                acc = wmma_bf16(a, bf, acc);
            }
            int d_ = tj * 16 + (lane & 15);
            if (d_ < 30) {
                int bi = w >> 8, wh = (w >> 4) & 15, ww = w & 15;
#pragma unroll
                for (int r = 0; r < 8; ++r) {
                    int nq = ti * 16 + r + ((lane >> 4) << 3);
                    int row = wh * 8 + (nq >> 3), col = ww * 8 + (nq & 7);
                    size_t tok = (size_t)bi * 16384 + row * 128 + col;
                    out[tok * C_DIM + h * 30 + d_] = f32_to_bf16(acc[r]);
                }
            }
        }
        __syncthreads();
    }
}

// ---------------------------------------------------------------------------
// launcher
// ---------------------------------------------------------------------------
extern "C" void kernel_launch(void* const* d_in, const int* in_sizes, int n_in,
                              void* d_out, int out_size, void* d_ws, size_t ws_size,
                              hipStream_t stream) {
    const float* x      = (const float*)d_in[0];
    const float* n1g    = (const float*)d_in[1];
    const float* n1b    = (const float*)d_in[2];
    const float* qkvw   = (const float*)d_in[3];
    const float* qkvb   = (const float*)d_in[4];
    const float* ppw    = (const float*)d_in[5];
    const float* ppb    = (const float*)d_in[6];
    const float* p1g    = (const float*)d_in[7];
    const float* p1b    = (const float*)d_in[8];
    const float* p1w    = (const float*)d_in[9];
    const float* p1bb   = (const float*)d_in[10];
    const float* p2g    = (const float*)d_in[11];
    const float* p2b    = (const float*)d_in[12];
    const float* p2w    = (const float*)d_in[13];
    const float* p2bb   = (const float*)d_in[14];
    const float* p3g    = (const float*)d_in[15];
    const float* p3b    = (const float*)d_in[16];
    const float* p3w    = (const float*)d_in[17];
    const float* p3bb   = (const float*)d_in[18];
    const float* projw  = (const float*)d_in[19];
    const float* projb  = (const float*)d_in[20];
    const float* n2g    = (const float*)d_in[21];
    const float* n2b    = (const float*)d_in[22];
    const float* fc1w   = (const float*)d_in[23];
    const float* fc1b   = (const float*)d_in[24];
    const float* fc2w   = (const float*)d_in[25];
    const float* fc2b   = (const float*)d_in[26];
    float* out = (float*)d_out;

    // workspace layout
    char* ws = (char*)d_ws;
    float* rpb  = (float*)ws;                              //  98304 B
    float* posT = (float*)(ws + 98304);                    //   8192 B slot
    u16* bufA   = (u16*)(ws + 98304 + 8192);               // 47.2 MB activations (bf16, C=180)
    u16* bufB   = (u16*)(ws + 98304 + 8192 +
                         (size_t)TOKENS * C_DIM * 2);      // 188.7 MB (qkv / ffn hidden)

    // 1) position-bias MLP + scatter
    pos_mlp_kernel<<<1, 256, 0, stream>>>(ppw, ppb, p1g, p1b, p1w, p1bb,
                                          p2g, p2b, p2w, p2bb,
                                          p3g, p3b, p3w, p3bb, posT);
    rpb_scatter_kernel<<<(6 * 4096 + 255) / 256, 256, 0, stream>>>(posT, rpb);

    // 2) LN1 + window partition -> bufA (bf16, window order)
    ln_kernel<<<TOKENS / 8, 256, 0, stream>>>(x, n1g, n1b, bufA, 1);

    // 3) QKV GEMM: [T,180] x [180,540] -> bufB bf16
    {
        int tiles = (TOKENS / 64) * ((QKV_DIM + 63) / 64);
        gemm_bf16<0><<<tiles, 256, 0, stream>>>(bufA, qkvw, qkvb, nullptr,
                                                bufB, nullptr,
                                                TOKENS, QKV_DIM, C_DIM);
    }

    // 4) window attention -> bufA (bf16, token order, window-reversed)
    attn_kernel<<<NWIN, 256, 0, stream>>>(bufB, rpb, bufA);

    // 5) proj GEMM + bias + residual(x) -> d_out (f32)   [x2 = shortcut + attn]
    {
        int tiles = (TOKENS / 64) * ((C_DIM + 63) / 64);
        gemm_bf16<2><<<tiles, 256, 0, stream>>>(bufA, projw, projb, x,
                                                nullptr, out,
                                                TOKENS, C_DIM, C_DIM);
    }

    // 6) LN2 -> bufA (bf16)
    ln_kernel<<<TOKENS / 8, 256, 0, stream>>>(out, n2g, n2b, bufA, 0);

    // 7) fc1 GEMM + bias + exact GELU -> bufB (bf16)
    {
        int tiles = (TOKENS / 64) * ((FF_DIM + 63) / 64);
        gemm_bf16<1><<<tiles, 256, 0, stream>>>(bufA, fc1w, fc1b, nullptr,
                                                bufB, nullptr,
                                                TOKENS, FF_DIM, C_DIM);
    }

    // 8) fc2 GEMM + bias + residual(x2 in d_out) -> d_out (f32, in-place RMW)
    {
        int tiles = (TOKENS / 64) * ((C_DIM + 63) / 64);
        gemm_bf16<2><<<tiles, 256, 0, stream>>>(bufB, fc2w, fc2b, out,
                                                nullptr, out,
                                                TOKENS, C_DIM, FF_DIM);
    }
}
